// MSAColumnAttentionV43_4861902979620
// MI455X (gfx1250) — matrix operand order repaired and checked by reading, not compiled, not run on hardware
//
#include <hip/hip_runtime.h>
#include <hip/hip_bf16.h>

// ---------------------------------------------------------------------------
// MSA column attention (B=1, S=256, N=384, C=512, H=8, Dh=64) for gfx1250.
// All GEMMs use v_wmma_f32_16x16x32_bf16 (fp32 accumulate). All fp32->bf16
// conversion is hoisted out of the GEMM inner loops.
// ---------------------------------------------------------------------------

#define HEADS  8
#define S_DIM  256
#define N_DIM  384
#define C_DIM  512
#define DH     64
#define NK     (N_DIM * DH)     // 24576 : contraction length for logits
#define C3     (3 * C_DIM)      // 1536
#define LN_EPS 1e-5f

typedef __attribute__((ext_vector_type(16))) __bf16 v16bf;
typedef __attribute__((ext_vector_type(8)))  __bf16 v8bf;
typedef __attribute__((ext_vector_type(8)))  float  v8f;
typedef __attribute__((ext_vector_type(4)))  float  v4f;
typedef __attribute__((ext_vector_type(2)))  unsigned v2u;

__device__ __forceinline__ __bf16 f2bf(float f) {
  unsigned u = __builtin_bit_cast(unsigned, f);
  unsigned r = (u + 0x7FFFu + ((u >> 16) & 1u)) >> 16;   // round-to-nearest-even
  unsigned short s = (unsigned short)r;
  return __builtin_bit_cast(__bf16, s);
}
__device__ __forceinline__ unsigned short bfbits(__bf16 b) {
  return __builtin_bit_cast(unsigned short, b);
}
__device__ __forceinline__ unsigned pack2bf(float a, float b) {
  return (unsigned)bfbits(f2bf(a)) | ((unsigned)bfbits(f2bf(b)) << 16);
}
__device__ __forceinline__ v16bf concat16(v8bf lo, v8bf hi) {
  return __builtin_shufflevector(lo, hi, 0, 1, 2, 3, 4, 5, 6, 7,
                                 8, 9, 10, 11, 12, 13, 14, 15);
}
__device__ __forceinline__ v16bf load16(const unsigned short* p) {
  return *(const v16bf*)p;                 // two global_load_b128
}
__device__ __forceinline__ v8bf load8(const unsigned short* p) {
  return *(const v8bf*)p;                  // one global_load_b128
}

// ---------------------------------------------------------------------------
// Kernel 1: convert Wqkv (1536x512) and Wout (512x512) fp32 -> bf16.
// ---------------------------------------------------------------------------
__global__ __launch_bounds__(256) void k_convert_weights(
    const float* __restrict__ Wqkv, const float* __restrict__ Wout,
    unsigned short* __restrict__ Wqkv_bf, unsigned short* __restrict__ Wout_bf) {
  int i = blockIdx.x * 256 + threadIdx.x;
  if (i < C3 * C_DIM) {
    Wqkv_bf[i] = bfbits(f2bf(Wqkv[i]));
  } else {
    int j = i - C3 * C_DIM;
    if (j < C_DIM * C_DIM) Wout_bf[j] = bfbits(f2bf(Wout[j]));
  }
}

// ---------------------------------------------------------------------------
// Kernel 2: convert msa (S*N*C fp32) -> bf16, 4 elements per thread.
// ---------------------------------------------------------------------------
__global__ __launch_bounds__(256) void k_convert_msa(
    const float* __restrict__ msa, unsigned short* __restrict__ msa_bf) {
  const size_t i = ((size_t)blockIdx.x * 256 + threadIdx.x) * 4;
  v4f f = *(const v4f*)(msa + i);
  v2u o;
  o[0] = pack2bf(f[0], f[1]);
  o[1] = pack2bf(f[2], f[3]);
  *(v2u*)(msa_bf + i) = o;
}

// ---------------------------------------------------------------------------
// Scatter helper for QKV epilogue: col j decides Q/K/V target + layout.
//   Q[h][s][n*Dh+dh], K[h][n*Dh+dh][t=s], Vt[n][h][dh][t=s]
// ---------------------------------------------------------------------------
__device__ __forceinline__ void scatter_qkv(
    const v8f& acc, int j, int n, int sbase, int half,
    const float* __restrict__ bqkv, unsigned short* __restrict__ Qb,
    unsigned short* __restrict__ Kb, unsigned short* __restrict__ Vt) {
  const float bias = bqkv[j];
  const int third = j / C_DIM;
  const int c0    = j % C_DIM;
  const int h     = c0 / DH;
  const int dh    = c0 % DH;
#pragma unroll
  for (int idx = 0; idx < 8; ++idx) {
    const int s = sbase + half * 8 + idx;
    const unsigned short bv = bfbits(f2bf(acc[idx] + bias));
    if (third == 0) {
      Qb[((size_t)h * S_DIM + s) * NK + n * DH + dh] = bv;
    } else if (third == 1) {
      Kb[(size_t)h * NK * S_DIM + (size_t)(n * DH + dh) * S_DIM + s] = bv;
    } else {
      Vt[((size_t)(n * HEADS + h) * DH + dh) * S_DIM + s] = bv;
    }
  }
}

// ---------------------------------------------------------------------------
// Kernel 3: QKV projection, bf16 in / bf16 out. One wave per 16x32 output
// tile (two WMMAs per A fragment), K-loop over C=512 in steps of 32.
// ---------------------------------------------------------------------------
__global__ __launch_bounds__(32) void k_qkv(
    const unsigned short* __restrict__ msa_bf, const float* __restrict__ bqkv,
    const unsigned short* __restrict__ Wqkv_bf,
    unsigned short* __restrict__ Qb, unsigned short* __restrict__ Kb,
    unsigned short* __restrict__ Vt) {
  const int lane  = threadIdx.x & 31;
  const int l     = lane & 15;
  const int half  = lane >> 4;
  const int jbase = blockIdx.x * 32;        // two 16-wide column tiles
  const int rbase = blockIdx.y * 16;        // r = n*S + s ; 16 | 256
  const int n     = rbase / S_DIM;
  const int sbase = rbase % S_DIM;

  const unsigned short* __restrict__ arow =
      msa_bf + ((size_t)(sbase + l) * N_DIM + n) * C_DIM;
  const unsigned short* __restrict__ brow0 =
      Wqkv_bf + (size_t)(jbase + l) * C_DIM;
  const unsigned short* __restrict__ brow1 =
      Wqkv_bf + (size_t)(jbase + 16 + l) * C_DIM;

  v8f acc0 = {}, acc1 = {};
#pragma unroll 4
  for (int kb = 0; kb < C_DIM; kb += 32) {
    v16bf a = concat16(load8(arow + kb + half * 16),
                       load8(arow + kb + half * 16 + 8));
    v16bf b0 = load16(brow0 + kb + half * 16);
    v16bf b1 = load16(brow1 + kb + half * 16);
    acc0 = __builtin_amdgcn_wmma_f32_16x16x32_bf16(false, a, false, b0,
                                                   (short)0, acc0, false, false);
    acc1 = __builtin_amdgcn_wmma_f32_16x16x32_bf16(false, a, false, b1,
                                                   (short)0, acc1, false, false);
  }
  scatter_qkv(acc0, jbase + l,      n, sbase, half, bqkv, Qb, Kb, Vt);
  scatter_qkv(acc1, jbase + 16 + l, n, sbase, half, bqkv, Qb, Kb, Vt);
}

// ---------------------------------------------------------------------------
// Kernel 4: attention logits. attn[h][s][t] = scale * sum_k Q[h][s][k]K[h][k][t]
// K-tile (32x16) staged through LDS (coalesced in t, ds_load_b128 out).
// 8 waves/block cover a 128(s) x 16(t) strip; K-loop = 768 WMMAs.
// ---------------------------------------------------------------------------
__global__ __launch_bounds__(256) void k_attn_logits(
    const unsigned short* __restrict__ Qb, const unsigned short* __restrict__ Kb,
    float* __restrict__ attnf) {
  __shared__ unsigned short bt[16][32];     // [t][k] transposed tile
  const int tid   = threadIdx.x;
  const int lane  = tid & 31;
  const int wave  = tid >> 5;
  const int l     = lane & 15;
  const int half  = lane >> 4;
  const int h     = blockIdx.z;
  const int tbase = blockIdx.x * 16;
  const int sbase = blockIdx.y * 128 + wave * 16;
  const int kk    = tid >> 3;               // 0..31 : k within tile
  const int dc    = tid & 7;                // 0..7  : dword (2 t values)

  const unsigned short* __restrict__ Kh = Kb + (size_t)h * NK * S_DIM;
  const unsigned short* __restrict__ arow =
      Qb + (size_t)h * S_DIM * NK + (size_t)(sbase + l) * NK;

  v8f acc = {};
  for (int kb = 0; kb < NK; kb += 32) {
    const unsigned kv =
        *(const unsigned*)(Kh + (size_t)(kb + kk) * S_DIM + tbase + dc * 2);
    __syncthreads();                        // previous tile fully consumed
    bt[dc * 2][kk]     = (unsigned short)(kv & 0xFFFFu);
    bt[dc * 2 + 1][kk] = (unsigned short)(kv >> 16);
    __syncthreads();

    v16bf a = concat16(load8(arow + kb + half * 16),
                       load8(arow + kb + half * 16 + 8));
    v16bf b = *(const v16bf*)&bt[l][half * 16];
    acc = __builtin_amdgcn_wmma_f32_16x16x32_bf16(false, a, false, b,
                                                  (short)0, acc, false, false);
  }

  const float scale = 0.125f;               // Dh^-0.5, Dh = 64
#pragma unroll
  for (int idx = 0; idx < 8; ++idx) {
    const int s = sbase + half * 8 + idx;
    attnf[((size_t)h * S_DIM + s) * S_DIM + tbase + l] = acc[idx] * scale;
  }
}

// ---------------------------------------------------------------------------
// Kernel 5: masked softmax over t (row length 256), one wave per row,
// bf16 output for the AV GEMM.
// ---------------------------------------------------------------------------
__global__ __launch_bounds__(256) void k_softmax(
    const float* __restrict__ attnf, const int* __restrict__ mask,
    unsigned short* __restrict__ attnb) {
  const int lane = threadIdx.x & 31;
  const int wave = threadIdx.x >> 5;
  const int row  = blockIdx.x * 8 + wave;   // h*S + s, 0..2047
  const float* __restrict__ in = attnf + (size_t)row * S_DIM;
  unsigned short* __restrict__ out = attnb + (size_t)row * S_DIM;

  float v[8];
  float mx = -3.0e38f;
#pragma unroll
  for (int j = 0; j < 8; ++j) {
    const int t = lane + j * 32;
    float x = in[t];
    if (mask[t] == 0) x = -1.0e9f;
    v[j] = x;
    mx = fmaxf(mx, x);
  }
#pragma unroll
  for (int off = 16; off > 0; off >>= 1) mx = fmaxf(mx, __shfl_xor(mx, off, 32));
  float sum = 0.f;
#pragma unroll
  for (int j = 0; j < 8; ++j) { v[j] = __expf(v[j] - mx); sum += v[j]; }
#pragma unroll
  for (int off = 16; off > 0; off >>= 1) sum += __shfl_xor(sum, off, 32);
  const float inv = 1.0f / sum;
#pragma unroll
  for (int j = 0; j < 8; ++j) out[lane + j * 32] = bfbits(f2bf(v[j] * inv));
}

// ---------------------------------------------------------------------------
// Kernel 6: out[n][s][h*64+dh] = sum_t attn[h][s][t] * Vt[n][h][dh][t].
// One wave per 16x16 tile; both fragments are contiguous bf16 loads.
// ---------------------------------------------------------------------------
__global__ __launch_bounds__(32) void k_av(
    const unsigned short* __restrict__ attnb, const unsigned short* __restrict__ Vt,
    unsigned short* __restrict__ outattn) {
  const int lane   = threadIdx.x & 31;
  const int l      = lane & 15;
  const int half   = lane >> 4;
  const int stile  = blockIdx.x >> 2;
  const int dtile  = blockIdx.x & 3;
  const int h      = blockIdx.y;
  const int n      = blockIdx.z;
  const int sbase  = stile * 16;
  const int dhbase = dtile * 16;

  const unsigned short* __restrict__ arow =
      attnb + ((size_t)h * S_DIM + sbase + l) * S_DIM;
  const unsigned short* __restrict__ brow =
      Vt + ((size_t)(n * HEADS + h) * DH + dhbase + l) * S_DIM;

  v8f acc = {};
#pragma unroll 2
  for (int kb = 0; kb < S_DIM; kb += 32) {
    v16bf a = concat16(load8(arow + kb + half * 16),
                       load8(arow + kb + half * 16 + 8));
    v16bf b = load16(brow + kb + half * 16);
    acc = __builtin_amdgcn_wmma_f32_16x16x32_bf16(false, a, false, b,
                                                  (short)0, acc, false, false);
  }
#pragma unroll
  for (int idx = 0; idx < 8; ++idx) {
    const int s = sbase + half * 8 + idx;
    outattn[((size_t)n * S_DIM + s) * C_DIM + h * DH + dhbase + l] =
        bfbits(f2bf(acc[idx]));
  }
}

// ---------------------------------------------------------------------------
// Kernel 7: output projection + bias + residual. Rows r = s*N + n (matches
// final (B,S,N,C) layout). One wave per 16x32 tile. fp32 x -> d_out.
// ---------------------------------------------------------------------------
__global__ __launch_bounds__(32) void k_outproj(
    const unsigned short* __restrict__ outattn,
    const unsigned short* __restrict__ Wout_bf,
    const float* __restrict__ bout, const float* __restrict__ msa,
    float* __restrict__ xout) {
  const int lane  = threadIdx.x & 31;
  const int l     = lane & 15;
  const int half  = lane >> 4;
  const int jbase = blockIdx.x * 32;
  const int rbase = blockIdx.y * 16;        // r = s*N + n ; 16 | 384
  const int s     = rbase / N_DIM;
  const int nbase = rbase % N_DIM;

  const unsigned short* __restrict__ arow =
      outattn + ((size_t)(nbase + l) * S_DIM + s) * C_DIM;
  const unsigned short* __restrict__ brow0 =
      Wout_bf + (size_t)(jbase + l) * C_DIM;
  const unsigned short* __restrict__ brow1 =
      Wout_bf + (size_t)(jbase + 16 + l) * C_DIM;

  v8f acc0 = {}, acc1 = {};
#pragma unroll 4
  for (int kb = 0; kb < C_DIM; kb += 32) {
    v16bf a = concat16(load8(arow + kb + half * 16),
                       load8(arow + kb + half * 16 + 8));
    v16bf b0 = load16(brow0 + kb + half * 16);
    v16bf b1 = load16(brow1 + kb + half * 16);
    acc0 = __builtin_amdgcn_wmma_f32_16x16x32_bf16(false, a, false, b0,
                                                   (short)0, acc0, false, false);
    acc1 = __builtin_amdgcn_wmma_f32_16x16x32_bf16(false, a, false, b1,
                                                   (short)0, acc1, false, false);
  }

#pragma unroll
  for (int g = 0; g < 2; ++g) {
    const int j = jbase + g * 16 + l;
    const float bias = bout[j];
    const v8f& acc = g ? acc1 : acc0;
#pragma unroll
    for (int idx = 0; idx < 8; ++idx) {
      const int nn = nbase + half * 8 + idx;
      const size_t addr = ((size_t)s * N_DIM + nn) * C_DIM + j;
      xout[addr] = acc[idx] + bias + msa[addr];
    }
  }
}

// ---------------------------------------------------------------------------
// Kernel 8: LayerNorm in place on d_out, one block per (s,n) row of 512.
// ---------------------------------------------------------------------------
__global__ __launch_bounds__(256) void k_layernorm(
    float* __restrict__ x, const float* __restrict__ gamma,
    const float* __restrict__ beta) {
  __shared__ float red[8];
  const int tid  = threadIdx.x;
  const int lane = tid & 31;
  const int wave = tid >> 5;
  float* __restrict__ row = x + (size_t)blockIdx.x * C_DIM;

  const float v0 = row[tid];
  const float v1 = row[tid + 256];
  float sum = v0 + v1;
#pragma unroll
  for (int off = 16; off > 0; off >>= 1) sum += __shfl_xor(sum, off, 32);
  if (lane == 0) red[wave] = sum;
  __syncthreads();
  float mean = 0.f;
#pragma unroll
  for (int i = 0; i < 8; ++i) mean += red[i];
  mean *= (1.0f / (float)C_DIM);
  __syncthreads();

  const float d0 = v0 - mean, d1 = v1 - mean;
  float vs = d0 * d0 + d1 * d1;
#pragma unroll
  for (int off = 16; off > 0; off >>= 1) vs += __shfl_xor(vs, off, 32);
  if (lane == 0) red[wave] = vs;
  __syncthreads();
  float var = 0.f;
#pragma unroll
  for (int i = 0; i < 8; ++i) var += red[i];
  var *= (1.0f / (float)C_DIM);
  const float rs = rsqrtf(var + LN_EPS);

  row[tid]       = d0 * rs * gamma[tid] + beta[tid];
  row[tid + 256] = d1 * rs * gamma[tid + 256] + beta[tid + 256];
}

// ---------------------------------------------------------------------------
// Launch. Workspace layout (bytes), total 407,896,064:
//   Wqkv_bf      0 ..   1,572,864
//   Wout_bf        .. 2,097,152
//   Qb             .. 102,760,448
//   Kb             .. 203,423,744
//   Vt             .. 304,087,040
//   attnf          .. 306,184,192
//   attnb          .. 307,232,768
//   msa_bf/outattn .. 407,896,064   (aliased: msa_bf dies after k_qkv,
//                                    outattn first written by k_av)
// ---------------------------------------------------------------------------
extern "C" void kernel_launch(void* const* d_in, const int* in_sizes, int n_in,
                              void* d_out, int out_size, void* d_ws, size_t ws_size,
                              hipStream_t stream) {
  (void)in_sizes; (void)n_in; (void)out_size; (void)ws_size;
  const float* msa   = (const float*)d_in[0];
  const int*   mask  = (const int*)d_in[1];
  const float* Wqkv  = (const float*)d_in[2];
  const float* bqkv  = (const float*)d_in[3];
  const float* Wout  = (const float*)d_in[4];
  const float* bout  = (const float*)d_in[5];
  const float* gamma = (const float*)d_in[6];
  const float* beta  = (const float*)d_in[7];
  float* out = (float*)d_out;
  char*  ws  = (char*)d_ws;

  unsigned short* Wqkv_bf = (unsigned short*)(ws);
  unsigned short* Wout_bf = (unsigned short*)(ws + 1572864);
  unsigned short* Qb      = (unsigned short*)(ws + 2097152);
  unsigned short* Kb      = (unsigned short*)(ws + 102760448);
  unsigned short* Vt      = (unsigned short*)(ws + 203423744);
  float*          attnf   = (float*)(ws + 304087040);
  unsigned short* attnb   = (unsigned short*)(ws + 306184192);
  unsigned short* msa_bf  = (unsigned short*)(ws + 307232768);
  unsigned short* outattn = msa_bf;   // aliased, non-overlapping live ranges

  k_convert_weights<<<4096, 256, 0, stream>>>(Wqkv, Wout, Wqkv_bf, Wout_bf);
  k_convert_msa<<<(S_DIM * N_DIM * C_DIM) / 1024, 256, 0, stream>>>(msa, msa_bf);
  k_qkv<<<dim3(C3 / 32, (N_DIM * S_DIM) / 16), 32, 0, stream>>>(
      msa_bf, bqkv, Wqkv_bf, Qb, Kb, Vt);
  k_attn_logits<<<dim3(S_DIM / 16, 2, HEADS), 256, 0, stream>>>(Qb, Kb, attnf);
  k_softmax<<<S_DIM, 256, 0, stream>>>(attnf, mask, attnb);
  k_av<<<dim3(64, HEADS, N_DIM), 32, 0, stream>>>(attnb, Vt, outattn);
  k_outproj<<<dim3(C_DIM / 32, (S_DIM * N_DIM) / 16), 32, 0, stream>>>(
      outattn, Wout_bf, bout, msa, out);
  k_layernorm<<<S_DIM * N_DIM, 256, 0, stream>>>(out, gamma, beta);
}
